// LongformerLayerWithMoE_82317343195590
// MI455X (gfx1250) — matrix operand order repaired and checked by reading, not compiled, not run on hardware
//
#include <hip/hip_runtime.h>
#include <math.h>

// ---------------------------------------------------------------------------
// Problem constants (from reference): B=1, S=4096, D=768, H=12, F=3072, E=7,
// TOPK=2, W=256. Head dim = 64, chunks = 16, window = 3W = 768.
// ---------------------------------------------------------------------------
#define S_LEN 4096
#define D_DIM 768
#define H_NUM 12
#define F_DIM 3072
#define E_NUM 7
#define W_WIN 256
#define DH    64
#define WIN   768            // 3*W
#define SPAD  (S_LEN + 2*W_WIN)   // 4608

typedef __bf16 bf16_t;
typedef __attribute__((ext_vector_type(16))) __bf16 v16bf;
typedef __attribute__((ext_vector_type(8)))  float  v8f;
typedef __attribute__((ext_vector_type(4)))  unsigned int uv4u;

// ---------------------------------------------------------------------------
// WMMA fragment helpers (layouts per CDNA5 ISA 7.12.2, wave32)
// A (16x32 bf16, row-major src): lane m = lane&15, K chunks [g*8, g*8+8) and
//   [g*8+16, g*8+24) with g = lane>>4  -> two 16B contiguous loads.
// B (32x16 bf16, stored as N x K row-major): lane n = lane&15,
//   K chunks [g*8, g*8+8) and [g*8+16, g*8+24) -> two 16B contiguous loads.
// C/D (16x16 f32): row = 8*(lane>>4)+r, col = lane&15.
// ---------------------------------------------------------------------------
__device__ inline v16bf load_frag_a(const bf16_t* __restrict__ base, int lda, int lane) {
    int m = lane & 15, g = lane >> 4;
    const bf16_t* p = base + (size_t)m * lda + g * 8;
    union { v16bf v; uv4u q[2]; } f;
    f.q[0] = *(const uv4u*)(p);
    f.q[1] = *(const uv4u*)(p + 16);
    return f.v;
}

__device__ inline v16bf load_frag_b(const bf16_t* __restrict__ base, int ldb, int lane) {
    int n = lane & 15, g = lane >> 4;
    const bf16_t* p = base + (size_t)n * ldb + g * 8;
    union { v16bf v; uv4u q[2]; } f;
    f.q[0] = *(const uv4u*)(p);
    f.q[1] = *(const uv4u*)(p + 16);
    return f.v;
}

__device__ inline v8f wmma_bf16(v16bf a, v16bf b, v8f c) {
    return __builtin_amdgcn_wmma_f32_16x16x32_bf16(false, a, false, b, (short)0, c,
                                                   false, false);
}

// ---------------------------------------------------------------------------
// Generic bf16 WMMA GEMM:  C(MxN) = A(MxK bf16 row-major) * Bt(NxK bf16)^T
// Block tile 64(M) x 256(N); 8 waves (2M x 4N), wave tile 32x64:
// 2 A frags x 4 B frags -> 8 WMMAs per 32-wide k-step, software pipelined
// (next step's 6 fragment loads are issued before the current WMMA burst).
// MODE 0: C = acc + bias                        (fp32 out)
// MODE 1: Cbf = bf16(gelu(acc + bias))          (expert hidden)
// MODE 2: C += rowscale[m*rs_stride]*(acc+bias) (MoE weighted accumulate)
// M % 64 == 0, N % 256 == 0, K % 32 == 0 (true for all uses here).
// ---------------------------------------------------------------------------
template <int MODE>
__global__ __launch_bounds__(256) void gemm_bf16_kernel(
    const bf16_t* __restrict__ A, const bf16_t* __restrict__ Bt,
    const float* __restrict__ bias, float* __restrict__ C,
    bf16_t* __restrict__ Cbf, const float* __restrict__ rowscale, int rs_stride,
    int M, int N, int K) {
    const int lane = threadIdx.x & 31;
    const int wave = threadIdx.x >> 5;
    const int wm = wave >> 2, wn = wave & 3;
    const int m0 = blockIdx.y * 64 + wm * 32;
    const int n0 = blockIdx.x * 256 + wn * 64;
    if (m0 + 32 > M || n0 + 64 > N) return;  // wave-uniform

    const bf16_t* Ap0 = A + (size_t)m0 * K;
    const bf16_t* Ap1 = Ap0 + (size_t)16 * K;
    const bf16_t* Bp0 = Bt + (size_t)(n0 + 0)  * K;
    const bf16_t* Bp1 = Bt + (size_t)(n0 + 16) * K;
    const bf16_t* Bp2 = Bt + (size_t)(n0 + 32) * K;
    const bf16_t* Bp3 = Bt + (size_t)(n0 + 48) * K;

    v8f acc[8] = {};
    v16bf a0 = load_frag_a(Ap0, K, lane);
    v16bf a1 = load_frag_a(Ap1, K, lane);
    v16bf b0 = load_frag_b(Bp0, K, lane);
    v16bf b1 = load_frag_b(Bp1, K, lane);
    v16bf b2 = load_frag_b(Bp2, K, lane);
    v16bf b3 = load_frag_b(Bp3, K, lane);

    for (int k0 = 0; k0 < K; k0 += 32) {
        v16bf na0 = a0, na1 = a1, nb0 = b0, nb1 = b1, nb2 = b2, nb3 = b3;
        const int kn = k0 + 32;
        if (kn < K) {  // prefetch next k-step before the WMMA burst
            na0 = load_frag_a(Ap0 + kn, K, lane);
            na1 = load_frag_a(Ap1 + kn, K, lane);
            nb0 = load_frag_b(Bp0 + kn, K, lane);
            nb1 = load_frag_b(Bp1 + kn, K, lane);
            nb2 = load_frag_b(Bp2 + kn, K, lane);
            nb3 = load_frag_b(Bp3 + kn, K, lane);
        }
        acc[0] = wmma_bf16(a0, b0, acc[0]);
        acc[1] = wmma_bf16(a1, b0, acc[1]);
        acc[2] = wmma_bf16(a0, b1, acc[2]);
        acc[3] = wmma_bf16(a1, b1, acc[3]);
        acc[4] = wmma_bf16(a0, b2, acc[4]);
        acc[5] = wmma_bf16(a1, b2, acc[5]);
        acc[6] = wmma_bf16(a0, b3, acc[6]);
        acc[7] = wmma_bf16(a1, b3, acc[7]);
        a0 = na0; a1 = na1; b0 = nb0; b1 = nb1; b2 = nb2; b3 = nb3;
    }

    const int mr = (lane >> 4) * 8, ncol = lane & 15;
#pragma unroll
    for (int t = 0; t < 4; ++t) {
#pragma unroll
        for (int half = 0; half < 2; ++half) {
#pragma unroll
            for (int r = 0; r < 8; ++r) {
                int m = m0 + half * 16 + mr + r;
                int n = n0 + t * 16 + ncol;
                float v = acc[t * 2 + half][r] + bias[n];
                if (MODE == 0) {
                    C[(size_t)m * N + n] = v;
                } else if (MODE == 1) {
                    float ge = 0.5f * v * (1.0f + erff(v * 0.70710678f));
                    Cbf[(size_t)m * N + n] = (bf16_t)ge;
                } else {
                    C[(size_t)m * N + n] += rowscale[(size_t)m * rs_stride] * v;
                }
            }
        }
    }
}

// ---------------------------------------------------------------------------
// Sliding-window attention, one block per (head, chunk, 16-query tile).
// Phase 1: scores = Q*K^T via WMMA into 48KB LDS (16x768 fp32); ALL fragments
//          (2 A + 12 B) are loaded into registers first, then one 12-WMMA
//          burst runs behind a single load wait.
// Phase 2: banded-mask softmax entirely in registers (48 elems/thread,
//          shfl_xor reductions); probs written back to LDS as bf16 (aliased).
// Phase 3: out = P*V via WMMA; A frags via ds_load_b128 from bf16 probs,
//          B frags from pre-transposed V; double-buffered over K=768.
// ---------------------------------------------------------------------------
__global__ __launch_bounds__(256) void attn_kernel(
    const bf16_t* __restrict__ qbf,   // [H][S][DH]   (pre-scaled by 1/8)
    const bf16_t* __restrict__ kpad,  // [H][SPAD][DH] (W zero-pad each side)
    const bf16_t* __restrict__ vt,    // [H][DH][SPAD]
    float* __restrict__ attn)         // [S][D]
{
    __shared__ alignas(16) float sc[16 * WIN];  // 48 KB; later aliased as bf16
    const int lane = threadIdx.x & 31, wave = threadIdx.x >> 5;
    const int qb = blockIdx.x & 15;
    const int c = (blockIdx.x >> 4) & 15;
    const int h = blockIdx.x >> 8;
    const int q0 = c * W_WIN + qb * 16;  // global query row of this tile

    // ---- Phase 1: scores (1 M-tile x 48 N-tiles; 6 N-tiles per wave) ----
    {
        const bf16_t* Q = qbf + ((size_t)h * S_LEN + q0) * DH;
        const bf16_t* Kb = kpad + ((size_t)h * SPAD + c * W_WIN) * DH;
        v16bf a0 = load_frag_a(Q, DH, lane);
        v16bf a1 = load_frag_a(Q + 32, DH, lane);
        v16bf bb[12];
#pragma unroll
        for (int t = 0; t < 6; ++t) {
            const bf16_t* kb = Kb + (size_t)((wave * 6 + t) * 16) * DH;
            bb[2 * t]     = load_frag_b(kb, DH, lane);
            bb[2 * t + 1] = load_frag_b(kb + 32, DH, lane);
        }
        v8f acc[6] = {};
#pragma unroll
        for (int t = 0; t < 6; ++t) {
            acc[t] = wmma_bf16(a0, bb[2 * t], acc[t]);
            acc[t] = wmma_bf16(a1, bb[2 * t + 1], acc[t]);
        }
        const int mr = (lane >> 4) * 8, ncol = lane & 15;
#pragma unroll
        for (int t = 0; t < 6; ++t)
#pragma unroll
            for (int r = 0; r < 8; ++r)
                sc[(mr + r) * WIN + (wave * 6 + t) * 16 + ncol] = acc[t][r];
    }
    __syncthreads();

    // ---- Phase 2: masked softmax over WIN=768 per row (regs only) ----
    {
        const int row = threadIdx.x >> 4, sub = threadIdx.x & 15;
        const int qi = qb * 16 + row;  // query index within chunk [0,256)
        const int jbase = sub * 48;
        float vals[48];
        float mx = -1e30f;
#pragma unroll
        for (int i = 0; i < 48; ++i) {
            int j = jbase + i;
            int pos = c * W_WIN + j - W_WIN;  // global key position
            bool ok = (j >= qi) && (j <= qi + 2 * W_WIN) && (pos >= 0) && (pos < S_LEN);
            float v = ok ? sc[row * WIN + j] : -1e9f;
            vals[i] = v;
            mx = fmaxf(mx, v);
        }
#pragma unroll
        for (int d = 8; d; d >>= 1) mx = fmaxf(mx, __shfl_xor(mx, d, 16));
        float sum = 0.f;
#pragma unroll
        for (int i = 0; i < 48; ++i) {
            vals[i] = __expf(vals[i] - mx);
            sum += vals[i];
        }
#pragma unroll
        for (int d = 8; d; d >>= 1) sum += __shfl_xor(sum, d, 16);
        float inv = 1.0f / sum;
        __syncthreads();  // all fp32 reads done before bf16 aliased writes
        bf16_t* pb = (bf16_t*)sc;
#pragma unroll
        for (int i = 0; i < 48; ++i)
            pb[row * WIN + jbase + i] = (bf16_t)(vals[i] * inv);
    }
    __syncthreads();

    // ---- Phase 3: P*V  (16 x 64 output; waves 0..3 take one 16x16 tile) ----
    if (wave < 4) {
        const bf16_t* Pb = (const bf16_t*)sc;  // [16][WIN] bf16 probs in LDS
        const bf16_t* Vb = vt + ((size_t)h * DH + wave * 16) * SPAD + c * W_WIN;
        v8f o = {};
        v16bf a = load_frag_a(Pb, WIN, lane);
        v16bf b = load_frag_b(Vb, SPAD, lane);
        for (int k0 = 0; k0 < WIN; k0 += 32) {
            v16bf na = a, nb = b;
            if (k0 + 32 < WIN) {
                na = load_frag_a(Pb + k0 + 32, WIN, lane);
                nb = load_frag_b(Vb + k0 + 32, SPAD, lane);
            }
            o = wmma_bf16(a, b, o);
            a = na; b = nb;
        }
        const int mr = (lane >> 4) * 8, ncol = lane & 15;
#pragma unroll
        for (int r = 0; r < 8; ++r)
            attn[(size_t)(q0 + mr + r) * D_DIM + h * DH + wave * 16 + ncol] = o[r];
    }
}

// ---------------------------------------------------------------------------
// LayerNorm over D=768: out = (x[+res] - mean)*rsqrt(var+eps)*g + b
// One block (256 threads) per row; also emits bf16 copy when requested.
// ---------------------------------------------------------------------------
__global__ __launch_bounds__(256) void ln_kernel(
    const float* __restrict__ x, const float* __restrict__ res,
    const float* __restrict__ g, const float* __restrict__ b,
    float* __restrict__ out, bf16_t* __restrict__ outbf) {
    const int row = blockIdx.x;
    const float* xr = x + (size_t)row * D_DIM;
    const float* rr = res ? res + (size_t)row * D_DIM : nullptr;
    float vals[3], lsum = 0.f, lsq = 0.f;
#pragma unroll
    for (int i = 0; i < 3; ++i) {
        int j = threadIdx.x + i * 256;
        float v = xr[j] + (rr ? rr[j] : 0.f);
        vals[i] = v;
        lsum += v;
        lsq += v * v;
    }
#pragma unroll
    for (int d = 16; d; d >>= 1) {
        lsum += __shfl_xor(lsum, d, 32);
        lsq += __shfl_xor(lsq, d, 32);
    }
    __shared__ float s1[8], s2[8], stats[2];
    if ((threadIdx.x & 31) == 0) {
        s1[threadIdx.x >> 5] = lsum;
        s2[threadIdx.x >> 5] = lsq;
    }
    __syncthreads();
    if (threadIdx.x == 0) {
        float ts = 0.f, tq = 0.f;
        for (int i = 0; i < 8; ++i) { ts += s1[i]; tq += s2[i]; }
        float mean = ts / (float)D_DIM;
        float var = tq / (float)D_DIM - mean * mean;
        stats[0] = mean;
        stats[1] = rsqrtf(var + 1e-5f);
    }
    __syncthreads();
    float mean = stats[0], inv = stats[1];
#pragma unroll
    for (int i = 0; i < 3; ++i) {
        int j = threadIdx.x + i * 256;
        float y = (vals[i] - mean) * inv * g[j] + b[j];
        out[(size_t)row * D_DIM + j] = y;
        if (outbf) outbf[(size_t)row * D_DIM + j] = (bf16_t)y;
    }
}

// ---------------------------------------------------------------------------
// Gate: logits (D x E dot), top-2, softmax over the two, scatter to [S][E].
// ---------------------------------------------------------------------------
__global__ __launch_bounds__(256) void gate_kernel(
    const float* __restrict__ ao, const float* __restrict__ gW,
    const float* __restrict__ gb, float* __restrict__ gout) {
    int s = blockIdx.x * 256 + threadIdx.x;
    if (s >= S_LEN) return;
    float logit[E_NUM];
#pragma unroll
    for (int e = 0; e < E_NUM; ++e) logit[e] = gb[e];
    const float* a = ao + (size_t)s * D_DIM;
    for (int j = 0; j < D_DIM; ++j) {
        float av = a[j];
#pragma unroll
        for (int e = 0; e < E_NUM; ++e) logit[e] += av * gW[j * E_NUM + e];
    }
    int i1 = 0; float v1 = logit[0];
#pragma unroll
    for (int e = 1; e < E_NUM; ++e)
        if (logit[e] > v1) { v1 = logit[e]; i1 = e; }
    int i2 = 0; float v2 = -1e30f;
#pragma unroll
    for (int e = 0; e < E_NUM; ++e)
        if (e != i1 && logit[e] > v2) { v2 = logit[e]; i2 = e; }
    float e2 = __expf(v2 - v1);
    float denom = 1.0f + e2;
#pragma unroll
    for (int e = 0; e < E_NUM; ++e) gout[(size_t)s * E_NUM + e] = 0.f;
    gout[(size_t)s * E_NUM + i1] = 1.0f / denom;
    gout[(size_t)s * E_NUM + i2] = e2 / denom;
}

// ---------------------------------------------------------------------------
// Utility kernels: fp32->bf16, batched KxN -> NxK bf16 transpose, zero,
// and QKV repacking (scale Q, zero-pad K, transpose+pad V).
// ---------------------------------------------------------------------------
__global__ void convert_bf16_kernel(const float* __restrict__ src,
                                    bf16_t* __restrict__ dst, long n) {
    long i = (long)blockIdx.x * 256 + threadIdx.x;
    if (i < n) dst[i] = (bf16_t)src[i];
}

__global__ void transpose_bf16_kernel(const float* __restrict__ src,
                                      bf16_t* __restrict__ dst, int nmat, int K,
                                      int N) {
    long idx = (long)blockIdx.x * 256 + threadIdx.x;
    long per = (long)K * N;
    long total = (long)nmat * per;
    if (idx >= total) return;
    int e = (int)(idx / per);
    long r = idx % per;
    int n = (int)(r / K), k = (int)(r % K);
    dst[(size_t)e * per + (size_t)n * K + k] =
        (bf16_t)src[(size_t)e * per + (size_t)k * N + n];
}

__global__ void zero_f32_kernel(float* __restrict__ p, long n) {
    long i = (long)blockIdx.x * 256 + threadIdx.x;
    if (i < n) p[i] = 0.f;
}

__global__ void pack_qkv_kernel(const float* __restrict__ q,
                                const float* __restrict__ k,
                                const float* __restrict__ v,
                                bf16_t* __restrict__ qbf,
                                bf16_t* __restrict__ kpad,
                                bf16_t* __restrict__ vt) {
    long idx = (long)blockIdx.x * 256 + threadIdx.x;
    long total = (long)H_NUM * SPAD * DH;
    if (idx >= total) return;
    int d = (int)(idx % DH);
    long rem = idx / DH;
    int sp = (int)(rem % SPAD);
    int h = (int)(rem / SPAD);
    int s = sp - W_WIN;
    bool in = (s >= 0) && (s < S_LEN);
    float kv = in ? k[(size_t)s * D_DIM + h * DH + d] : 0.f;
    float vv = in ? v[(size_t)s * D_DIM + h * DH + d] : 0.f;
    kpad[((size_t)h * SPAD + sp) * DH + d] = (bf16_t)kv;
    vt[((size_t)h * DH + d) * SPAD + sp] = (bf16_t)vv;
    if (in)
        qbf[((size_t)h * S_LEN + s) * DH + d] =
            (bf16_t)(q[(size_t)s * D_DIM + h * DH + d] * 0.125f);
}

// ---------------------------------------------------------------------------
// Host orchestration
// ---------------------------------------------------------------------------
static inline dim3 grid1(long n) { return dim3((unsigned)((n + 255) / 256)); }

extern "C" void kernel_launch(void* const* d_in, const int* in_sizes, int n_in,
                              void* d_out, int out_size, void* d_ws,
                              size_t ws_size, hipStream_t stream) {
    (void)in_sizes; (void)n_in; (void)out_size; (void)ws_size;
    const float* x     = (const float*)d_in[0];
    const float* Wq    = (const float*)d_in[1];
    const float* bq    = (const float*)d_in[2];
    const float* Wk    = (const float*)d_in[3];
    const float* bk    = (const float*)d_in[4];
    const float* Wv    = (const float*)d_in[5];
    const float* bv    = (const float*)d_in[6];
    const float* Wo    = (const float*)d_in[7];
    const float* bo    = (const float*)d_in[8];
    const float* ln1g  = (const float*)d_in[9];
    const float* ln1b  = (const float*)d_in[10];
    const float* gateW = (const float*)d_in[11];
    const float* gateb = (const float*)d_in[12];
    const float* W1e   = (const float*)d_in[13];
    const float* b1e   = (const float*)d_in[14];
    const float* W2e   = (const float*)d_in[15];
    const float* b2e   = (const float*)d_in[16];
    const float* Wd    = (const float*)d_in[17];
    const float* bd    = (const float*)d_in[18];
    const float* ln2g  = (const float*)d_in[19];
    const float* ln2b  = (const float*)d_in[20];

    float* out_layer = (float*)d_out;                          // [S][D]
    float* out_gate  = (float*)d_out + (size_t)S_LEN * D_DIM;  // [S][E]

    // ---- workspace carve-out ----
    char* ws = (char*)d_ws;
    size_t off = 0;
    auto alloc = [&](size_t bytes) -> char* {
        char* p = ws + off;
        off += (bytes + 255) & ~(size_t)255;
        return p;
    };
    bf16_t* x_bf  = (bf16_t*)alloc((size_t)S_LEN * D_DIM * 2);
    bf16_t* wqT   = (bf16_t*)alloc((size_t)D_DIM * D_DIM * 2);
    bf16_t* wkT   = (bf16_t*)alloc((size_t)D_DIM * D_DIM * 2);
    bf16_t* wvT   = (bf16_t*)alloc((size_t)D_DIM * D_DIM * 2);
    bf16_t* woT   = (bf16_t*)alloc((size_t)D_DIM * D_DIM * 2);
    bf16_t* wdT   = (bf16_t*)alloc((size_t)D_DIM * D_DIM * 2);
    bf16_t* w1T   = (bf16_t*)alloc((size_t)E_NUM * F_DIM * D_DIM * 2);
    bf16_t* w2T   = (bf16_t*)alloc((size_t)E_NUM * D_DIM * F_DIM * 2);
    float*  qf    = (float*)alloc((size_t)S_LEN * D_DIM * 4);
    float*  kf    = (float*)alloc((size_t)S_LEN * D_DIM * 4);
    float*  vf    = (float*)alloc((size_t)S_LEN * D_DIM * 4);
    bf16_t* qbf   = (bf16_t*)alloc((size_t)H_NUM * S_LEN * DH * 2);
    bf16_t* kpad  = (bf16_t*)alloc((size_t)H_NUM * SPAD * DH * 2);
    bf16_t* vt    = (bf16_t*)alloc((size_t)H_NUM * DH * SPAD * 2);
    float*  attn  = (float*)alloc((size_t)S_LEN * D_DIM * 4);
    bf16_t* attnb = (bf16_t*)alloc((size_t)S_LEN * D_DIM * 2);
    float*  woout = (float*)alloc((size_t)S_LEN * D_DIM * 4);
    float*  ao    = (float*)alloc((size_t)S_LEN * D_DIM * 4);
    bf16_t* aobf  = (bf16_t*)alloc((size_t)S_LEN * D_DIM * 2);
    bf16_t* hbuf  = (bf16_t*)alloc((size_t)S_LEN * F_DIM * 2);
    float*  moe   = (float*)alloc((size_t)S_LEN * D_DIM * 4);
    bf16_t* moebf = (bf16_t*)alloc((size_t)S_LEN * D_DIM * 2);
    float*  ffn   = (float*)alloc((size_t)S_LEN * D_DIM * 4);

    const dim3 blk(256);
    const dim3 gemmD(D_DIM / 256, S_LEN / 64);  // N=768 GEMMs
    const dim3 gemmF(F_DIM / 256, S_LEN / 64);  // N=3072 GEMMs

    // 1) bf16 conversions / transposes of weights + activations
    convert_bf16_kernel<<<grid1((long)S_LEN * D_DIM), blk, 0, stream>>>(
        x, x_bf, (long)S_LEN * D_DIM);
    transpose_bf16_kernel<<<grid1((long)D_DIM * D_DIM), blk, 0, stream>>>(Wq, wqT, 1, D_DIM, D_DIM);
    transpose_bf16_kernel<<<grid1((long)D_DIM * D_DIM), blk, 0, stream>>>(Wk, wkT, 1, D_DIM, D_DIM);
    transpose_bf16_kernel<<<grid1((long)D_DIM * D_DIM), blk, 0, stream>>>(Wv, wvT, 1, D_DIM, D_DIM);
    transpose_bf16_kernel<<<grid1((long)D_DIM * D_DIM), blk, 0, stream>>>(Wo, woT, 1, D_DIM, D_DIM);
    transpose_bf16_kernel<<<grid1((long)D_DIM * D_DIM), blk, 0, stream>>>(Wd, wdT, 1, D_DIM, D_DIM);
    transpose_bf16_kernel<<<grid1((long)E_NUM * D_DIM * F_DIM), blk, 0, stream>>>(
        W1e, w1T, E_NUM, D_DIM, F_DIM);
    transpose_bf16_kernel<<<grid1((long)E_NUM * F_DIM * D_DIM), blk, 0, stream>>>(
        W2e, w2T, E_NUM, F_DIM, D_DIM);

    // 2) Q/K/V projections (WMMA GEMM, fp32 out + bias)
    gemm_bf16_kernel<0><<<gemmD, blk, 0, stream>>>(x_bf, wqT, bq, qf, nullptr, nullptr, 0,
                                                   S_LEN, D_DIM, D_DIM);
    gemm_bf16_kernel<0><<<gemmD, blk, 0, stream>>>(x_bf, wkT, bk, kf, nullptr, nullptr, 0,
                                                   S_LEN, D_DIM, D_DIM);
    gemm_bf16_kernel<0><<<gemmD, blk, 0, stream>>>(x_bf, wvT, bv, vf, nullptr, nullptr, 0,
                                                   S_LEN, D_DIM, D_DIM);

    // 3) repack to per-head bf16 (scaled Q, padded K, transposed padded V)
    pack_qkv_kernel<<<grid1((long)H_NUM * SPAD * DH), blk, 0, stream>>>(qf, kf, vf, qbf,
                                                                        kpad, vt);

    // 4) sliding-window attention (WMMA QK^T, softmax, WMMA PV)
    attn_kernel<<<dim3(H_NUM * 16 * 16), blk, 0, stream>>>(qbf, kpad, vt, attn);

    // 5) output projection + residual + LN1 (also emits bf16 activations)
    convert_bf16_kernel<<<grid1((long)S_LEN * D_DIM), blk, 0, stream>>>(
        attn, attnb, (long)S_LEN * D_DIM);
    gemm_bf16_kernel<0><<<gemmD, blk, 0, stream>>>(attnb, woT, bo, woout, nullptr,
                                                   nullptr, 0, S_LEN, D_DIM, D_DIM);
    ln_kernel<<<dim3(S_LEN), blk, 0, stream>>>(woout, x, ln1g, ln1b, ao, aobf);

    // 6) router: logits + top-2 softmax -> gate_weights (second output)
    gate_kernel<<<dim3(S_LEN / 256), blk, 0, stream>>>(ao, gateW, gateb, out_gate);

    // 7) MoE: per-expert GEMM1(gelu->bf16) then GEMM2 accumulated with gate wt
    zero_f32_kernel<<<grid1((long)S_LEN * D_DIM), blk, 0, stream>>>(moe,
                                                                    (long)S_LEN * D_DIM);
    for (int e = 0; e < E_NUM; ++e) {
        gemm_bf16_kernel<1><<<gemmF, blk, 0, stream>>>(
            aobf, w1T + (size_t)e * F_DIM * D_DIM, b1e + (size_t)e * F_DIM, nullptr,
            hbuf, nullptr, 0, S_LEN, F_DIM, D_DIM);
        gemm_bf16_kernel<2><<<gemmD, blk, 0, stream>>>(
            hbuf, w2T + (size_t)e * D_DIM * F_DIM, b2e + (size_t)e * D_DIM, moe,
            nullptr, out_gate + e, E_NUM, S_LEN, D_DIM, F_DIM);
    }

    // 8) dense projection + residual + LN2 -> layer output
    convert_bf16_kernel<<<grid1((long)S_LEN * D_DIM), blk, 0, stream>>>(
        moe, moebf, (long)S_LEN * D_DIM);
    gemm_bf16_kernel<0><<<gemmD, blk, 0, stream>>>(moebf, wdT, bd, ffn, nullptr, nullptr,
                                                   0, S_LEN, D_DIM, D_DIM);
    ln_kernel<<<dim3(S_LEN), blk, 0, stream>>>(ffn, ao, ln2g, ln2b, out_layer, nullptr);
}
